// AttributeGNN_43284680409242
// MI455X (gfx1250) — matrix operand order repaired and checked by reading, compiled, not run on hardware
//
#include <hip/hip_runtime.h>

// ---------------- constants ----------------
#define B_ROWS 16384
#define A_DIM  8
#define E_DIM  512
#define K2E    1024
#define ROWS   32              // b-rows per block
#define THREADS 512            // 16 wave32s: 2 M-halves x 8 N-groups(64 cols)

typedef __attribute__((ext_vector_type(16))) __bf16          v16bf;
typedef __attribute__((ext_vector_type(16))) unsigned short  v16u;
typedef __attribute__((ext_vector_type(8)))  unsigned short  v8u;
typedef __attribute__((ext_vector_type(8)))  float           v8f;

struct V8Pair { v8u lo; v8u hi; };   // 32B: two halves of a bf16 fragment

__device__ __forceinline__ unsigned short f2bf(float f) {
    unsigned int u = __builtin_bit_cast(unsigned int, f);
    u += 0x7FFFu + ((u >> 16) & 1u);            // round-to-nearest-even
    return (unsigned short)(u >> 16);
}

// XOR-swizzled LDS layout, pitch 512 shorts (no padding, exactly 64KB for 2 bufs).
// Swizzle permutes 8-short (16B) chunks: ds_load_b128 runs stay contiguous, and the
// 16 rows of an A-fragment map to distinct bank groups -> conflict-free.
__device__ __forceinline__ int sw_idx(int row, int col) {
    return row * E_DIM + (col ^ ((row & 15) << 3));
}

// A-fragment (16x32 bf16) in contiguous k order (not the ISA table order):
// lanes 0-15: M=lane, K=k0..k0+15; lanes 16-31: K=k0+16..k0+31 (matches B-fragment).
__device__ __forceinline__ v16bf lds_afrag(const unsigned short* s, int rowBaseL,
                                           int lane, int k0) {
    const int row = rowBaseL + (lane & 15);
    const int kb  = k0 + ((lane >> 4) << 4);
    const int swz = (row & 15) << 3;
    V8Pair t;
    t.lo = *(const v8u*)(s + row * E_DIM + (kb ^ swz));
    t.hi = *(const v8u*)(s + row * E_DIM + ((kb + 8) ^ swz));
    return __builtin_bit_cast(v16bf, t);
}

// B-fragment (32x16 bf16) from prepacked bf16 weights [N][K]:
// lane%16 = N, lanes 0-15 hold K=k0..k0+15, lanes 16-31 K=k0+16..k0+31.
// 32B contiguous per lane -> 2x global_load_b128 (saddr + 32-bit offset).
__device__ __forceinline__ v16bf g_bfrag_packed(const unsigned short* g, int ldk,
                                                int n0, int k0, int lane) {
    const int off = (n0 + (lane & 15)) * ldk + k0 + ((lane >> 4) << 4);
    v16u bv = *(const v16u*)(g + off);
    return __builtin_bit_cast(v16bf, bv);
}

// Fallback: raw f32, B[k][n] = w[n*ldk + k]  (linear_w layout)
__device__ __forceinline__ v16bf g_bfrag_rawT(const float* w, int ldk,
                                              int n0, int k0, int lane) {
    const float* p = w + (size_t)(n0 + (lane & 15)) * ldk + k0 + ((lane >> 4) << 4);
    v16u bv;
#pragma unroll
    for (int i = 0; i < 16; ++i) bv[i] = f2bf(p[i]);
    return __builtin_bit_cast(v16bf, bv);
}

// Fallback: raw f32, B[k][n] = w[k*ld + n]  (fwd/bwd_proj layout)
__device__ __forceinline__ v16bf g_bfrag_rawN(const float* w, int ld,
                                              int n0, int k0, int lane) {
    const int col = n0 + (lane & 15);
    const int kb  = k0 + ((lane >> 4) << 4);
    v16u bv;
#pragma unroll
    for (int i = 0; i < 16; ++i) bv[i] = f2bf(w[(size_t)(kb + i) * ld + col]);
    return __builtin_bit_cast(v16bf, bv);
}

__device__ __forceinline__ v8f wmma_bf16(v16bf a, v16bf b, v8f c) {
    return __builtin_amdgcn_wmma_f32_16x16x32_bf16(false, a, false, b, (short)0, c, false, false);
}

// Stage a ROWSx512 f32 panel into swizzled bf16 LDS (float4 loads, b64 stores).
__device__ __forceinline__ void stagePanel(const float* __restrict__ g, size_t rowStride,
                                           unsigned short* s, int tid) {
#pragma unroll
    for (int i = 0; i < ROWS * 128 / THREADS; ++i) {
        const int c    = tid + i * THREADS;       // ROWS*128 float4 chunks
        const int row  = c >> 7;
        const int col  = (c & 127) * 4;
        const float4 v = *(const float4*)(g + (size_t)row * rowStride + col);
        unsigned long long pk = (unsigned long long)f2bf(v.x)
                              | ((unsigned long long)f2bf(v.y) << 16)
                              | ((unsigned long long)f2bf(v.z) << 32)
                              | ((unsigned long long)f2bf(v.w) << 48);
        *(unsigned long long*)(s + sw_idx(row, col)) = pk;
    }
}

// ---------------- prepack kernels (f32 -> bf16; proj transposed to [a][j][i]) --------
__global__ void pack_linw_k(const float* __restrict__ w, unsigned short* __restrict__ o, int n4) {
    int i = blockIdx.x * blockDim.x + threadIdx.x;
    if (i >= n4) return;
    float4 v = ((const float4*)w)[i];
    unsigned long long pk = (unsigned long long)f2bf(v.x)
                          | ((unsigned long long)f2bf(v.y) << 16)
                          | ((unsigned long long)f2bf(v.z) << 32)
                          | ((unsigned long long)f2bf(v.w) << 48);
    ((unsigned long long*)o)[i] = pk;
}

__global__ void pack_projT_k(const float* __restrict__ src, unsigned short* __restrict__ dst) {
    __shared__ unsigned short tile[32][33];
    int a  = blockIdx.z;
    int i0 = blockIdx.x * 32, j0 = blockIdx.y * 32;
    const float* s    = src + (size_t)a * E_DIM * E_DIM;
    unsigned short* d = dst + (size_t)a * E_DIM * E_DIM;
    for (int r = threadIdx.y; r < 32; r += 8)
        tile[r][threadIdx.x] = f2bf(s[(size_t)(i0 + r) * E_DIM + j0 + threadIdx.x]);
    __syncthreads();
    for (int r = threadIdx.y; r < 32; r += 8)
        d[(size_t)(j0 + r) * E_DIM + i0 + threadIdx.x] = tile[threadIdx.x][r];
}

// ---------------- fused GNN kernel ----------------
template <bool PACKED>
__global__ __launch_bounds__(THREADS) void fused_attr_gnn(
    const float* __restrict__ img,  const float* __restrict__ edge,
    const float* __restrict__ linW, const float* __restrict__ linB,
    const float* __restrict__ fwdP, const float* __restrict__ bwdP,
    const float* __restrict__ sw,
    const unsigned short* __restrict__ Wp, const unsigned short* __restrict__ Fp,
    const unsigned short* __restrict__ Bp,
    float* __restrict__ attrOut, float* __restrict__ indOut)
{
    __shared__ unsigned short sA[ROWS * E_DIM];   // edge -> agg      (32 KB)
    __shared__ unsigned short sB[ROWS * E_DIM];   // src  -> attr_rep (32 KB)

    const int tid   = threadIdx.x;
    const int lane  = tid & 31;
    const int wave  = tid >> 5;                   // 0..15
    const int mBase = (wave >> 3) << 4;           // 0 or 16 (row half)
    const int nBase = (wave & 7) * 64;            // 8 groups x 64 cols
    const int rowBase = blockIdx.x * ROWS;
    const int mOff  = (lane >> 4) << 3;           // C-fragment: VGPR r -> row r+mOff
    const int nLane = lane & 15;

    stagePanel(img + (size_t)rowBase * E_DIM, E_DIM, sB, tid);

    v8f accInd[4];
#pragma unroll
    for (int t = 0; t < 4; ++t)
#pragma unroll
        for (int r = 0; r < 8; ++r) accInd[t][r] = 0.f;

    __syncthreads();

    // cBase = bias + src @ W1^T  (identical for all 8 attributes -> compute once).
    // k-outer / t-inner: one A-fragment feeds 4 independent WMMA chains.
    v8f cBase[4];
#pragma unroll
    for (int t = 0; t < 4; ++t) {
        const float bias = linB[nBase + t * 16 + nLane];
#pragma unroll
        for (int r = 0; r < 8; ++r) cBase[t][r] = bias;
    }
#pragma unroll 2
    for (int k0 = 0; k0 < E_DIM; k0 += 32) {
        const v16bf aS = lds_afrag(sB, mBase, lane, k0);
#pragma unroll
        for (int t = 0; t < 4; ++t) {
            const v16bf b = PACKED ? g_bfrag_packed(Wp, K2E, nBase + t * 16, k0, lane)
                                   : g_bfrag_rawT(linW, K2E, nBase + t * 16, k0, lane);
            cBase[t] = wmma_bf16(aS, b, cBase[t]);
        }
    }

    for (int a = 0; a < A_DIM; ++a) {
        stagePanel(edge + ((size_t)rowBase * A_DIM + a) * E_DIM,
                   (size_t)A_DIM * E_DIM, sA, tid);
        __syncthreads();                          // B1: edge visible

        // Warm L2/L0 for the next attribute's edge panel while we compute.
        if (a + 1 < A_DIM) {
            const float* nxt = edge + ((size_t)rowBase * A_DIM + (a + 1)) * E_DIM;
            __builtin_prefetch(nxt + tid * 32, 0, 1);        // 512 thr x 128B = 64KB panel
            __builtin_prefetch(nxt + tid * 32 + 16, 0, 1);
        }

        const float swa = sw[a];

        // GEMM1 (edge half): agg = cBase + edge @ W2^T
        v8f cAgg[4];
#pragma unroll
        for (int t = 0; t < 4; ++t) cAgg[t] = cBase[t];
#pragma unroll 2
        for (int k0 = 0; k0 < E_DIM; k0 += 32) {
            const v16bf aE = lds_afrag(sA, mBase, lane, k0);
#pragma unroll
            for (int t = 0; t < 4; ++t) {
                const v16bf b = PACKED ? g_bfrag_packed(Wp, K2E, nBase + t * 16, E_DIM + k0, lane)
                                       : g_bfrag_rawT(linW, K2E, nBase + t * 16, E_DIM + k0, lane);
                cAgg[t] = wmma_bf16(aE, b, cAgg[t]);
            }
        }
        __syncthreads();                          // B2: all reads of sA done

        // agg -> bf16 into sA (each wave owns its 64 cols of its row half)
#pragma unroll
        for (int t = 0; t < 4; ++t) {
#pragma unroll
            for (int r = 0; r < 8; ++r)
                sA[sw_idx(mBase + mOff + r, nBase + t * 16 + nLane)] = f2bf(cAgg[t][r]);
        }
        __syncthreads();                          // B3: agg visible

        // GEMM2: attr_rep = agg @ fwd_proj[a]
        const unsigned short* Fpa  = Fp   + (size_t)a * E_DIM * E_DIM;
        const float*          fwda = fwdP + (size_t)a * E_DIM * E_DIM;
        v8f cRep[4];
#pragma unroll
        for (int t = 0; t < 4; ++t)
#pragma unroll
            for (int r = 0; r < 8; ++r) cRep[t][r] = 0.f;
#pragma unroll 2
        for (int k0 = 0; k0 < E_DIM; k0 += 32) {
            const v16bf aF = lds_afrag(sA, mBase, lane, k0);
#pragma unroll
            for (int t = 0; t < 4; ++t) {
                const v16bf b = PACKED ? g_bfrag_packed(Fpa, E_DIM, nBase + t * 16, k0, lane)
                                       : g_bfrag_rawN(fwda, E_DIM, nBase + t * 16, k0, lane);
                cRep[t] = wmma_bf16(aF, b, cRep[t]);
            }
        }
        // store attr_rep (f32) and stage it as bf16 into sB
#pragma unroll
        for (int t = 0; t < 4; ++t) {
#pragma unroll
            for (int r = 0; r < 8; ++r) {
                const size_t gRow = rowBase + mBase + mOff + r;
                attrOut[(gRow * A_DIM + a) * E_DIM + nBase + t * 16 + nLane] = cRep[t][r];
                sB[sw_idx(mBase + mOff + r, nBase + t * 16 + nLane)] = f2bf(cRep[t][r]);
            }
        }
        __syncthreads();                          // B4: rep visible; sA reads done

        // GEMM3: proj = attr_rep @ bwd_proj[a]; individuals += relu(proj)*sw[a]
        const unsigned short* Bpa  = Bp   + (size_t)a * E_DIM * E_DIM;
        const float*          bwda = bwdP + (size_t)a * E_DIM * E_DIM;
        v8f cPrj[4];
#pragma unroll
        for (int t = 0; t < 4; ++t)
#pragma unroll
            for (int r = 0; r < 8; ++r) cPrj[t][r] = 0.f;
#pragma unroll 2
        for (int k0 = 0; k0 < E_DIM; k0 += 32) {
            const v16bf aR = lds_afrag(sB, mBase, lane, k0);
#pragma unroll
            for (int t = 0; t < 4; ++t) {
                const v16bf b = PACKED ? g_bfrag_packed(Bpa, E_DIM, nBase + t * 16, k0, lane)
                                       : g_bfrag_rawN(bwda, E_DIM, nBase + t * 16, k0, lane);
                cPrj[t] = wmma_bf16(aR, b, cPrj[t]);
            }
        }
#pragma unroll
        for (int t = 0; t < 4; ++t)
#pragma unroll
            for (int r = 0; r < 8; ++r)
                accInd[t][r] += fmaxf(cPrj[t][r], 0.f) * swa;
        // no trailing barrier needed: next edge-stage (sA) is fenced by B4,
        // next rep-write (sB) is fenced by next iteration's B1..B3.
    }

    // individuals
#pragma unroll
    for (int t = 0; t < 4; ++t) {
#pragma unroll
        for (int r = 0; r < 8; ++r) {
            const size_t gRow = rowBase + mBase + mOff + r;
            indOut[gRow * E_DIM + nBase + t * 16 + nLane] = accInd[t][r];
        }
    }
}

// ---------------- host ----------------
extern "C" void kernel_launch(void* const* d_in, const int* in_sizes, int n_in,
                              void* d_out, int out_size, void* d_ws, size_t ws_size,
                              hipStream_t stream) {
    (void)in_sizes; (void)n_in; (void)out_size;
    const float* img  = (const float*)d_in[0];
    const float* edge = (const float*)d_in[1];
    const float* linW = (const float*)d_in[2];
    const float* linB = (const float*)d_in[3];
    const float* fwdP = (const float*)d_in[4];
    const float* bwdP = (const float*)d_in[5];
    const float* sw   = (const float*)d_in[6];

    float* attrOut = (float*)d_out;
    float* indOut  = attrOut + (size_t)B_ROWS * A_DIM * E_DIM;

    unsigned short* Wp = (unsigned short*)d_ws;                 // [512][1024] bf16
    unsigned short* Fp = Wp + (size_t)E_DIM * K2E;              // [8][j][i] bf16
    unsigned short* Bp = Fp + (size_t)A_DIM * E_DIM * E_DIM;    // [8][j][i] bf16
    const size_t need = ((size_t)E_DIM * K2E + 2 * (size_t)A_DIM * E_DIM * E_DIM)
                        * sizeof(unsigned short);               // ~9 MB

    if (ws_size >= need) {
        pack_linw_k<<<dim3((E_DIM * K2E / 4 + 255) / 256), dim3(256), 0, stream>>>(
            linW, Wp, E_DIM * K2E / 4);
        pack_projT_k<<<dim3(16, 16, A_DIM), dim3(32, 8), 0, stream>>>(fwdP, Fp);
        pack_projT_k<<<dim3(16, 16, A_DIM), dim3(32, 8), 0, stream>>>(bwdP, Bp);
        fused_attr_gnn<true><<<dim3(B_ROWS / ROWS), dim3(THREADS), 0, stream>>>(
            img, edge, linW, linB, fwdP, bwdP, sw, Wp, Fp, Bp, attrOut, indOut);
    } else {
        fused_attr_gnn<false><<<dim3(B_ROWS / ROWS), dim3(THREADS), 0, stream>>>(
            img, edge, linW, linB, fwdP, bwdP, sw, Wp, Fp, Bp, attrOut, indOut);
    }
}
